// DownLayer_8083128451159
// MI455X (gfx1250) — matrix operand order repaired and checked by reading, not compiled, not run on hardware
//
#include <hip/hip_runtime.h>
#include <hip/hip_bf16.h>

typedef __attribute__((ext_vector_type(16))) _Float16 v16h;
typedef __attribute__((ext_vector_type(8)))  _Float16 v8h;
typedef __attribute__((ext_vector_type(8)))  float    v8f;

#define B_    8
#define N_    4096
#define S_    1024
#define NS_   32
#define CIN_  64
#define CMLP_ 128
#define COUT_ 256
#define RAD2_ (0.15f * 0.15f)
#define NSAMPLES_TOTAL (B_ * S_ * NS_)   // 262144
#define LDSA_STRIDE 136                  // 128 + 8 f16 pad (272 B, 16B-aligned)

// ---------------------------------------------------------------------------
// K0: zero the BN accumulators (512 floats)
// ---------------------------------------------------------------------------
__global__ void init_stats_kernel(float* stats) {
    int t = threadIdx.x;
    if (t < 512) stats[t] = 0.0f;
}

// ---------------------------------------------------------------------------
// K0b: one-time weight prep: f32 row-major -> f16 column-major (transposed),
// so WMMA B-fragments become contiguous 32-byte vector loads.
//   W1T[col*64  + k] = W1[k*128 + col]   (128 cols x 64 K)
//   WcT[col*128 + k] = Wc[k*256 + col]   (256 cols x 128 K)
// ---------------------------------------------------------------------------
__global__ void prep_weights_kernel(const float* __restrict__ W1,
                                    const float* __restrict__ Wc,
                                    _Float16* __restrict__ W1T,
                                    _Float16* __restrict__ WcT) {
    int t = blockIdx.x * blockDim.x + threadIdx.x;
    if (t < CMLP_ * CIN_) {
        int col = t / CIN_, k = t % CIN_;
        W1T[t] = (_Float16)W1[k * CMLP_ + col];
    }
    if (t < COUT_ * CMLP_) {
        int col = t / CMLP_, k = t % CMLP_;
        WcT[t] = (_Float16)Wc[k * COUT_ + col];
    }
}

// ---------------------------------------------------------------------------
// K1: farthest point sampling. One block (1024 thr = 32 waves) per batch.
// ---------------------------------------------------------------------------
__global__ __launch_bounds__(1024) void fps_kernel(const float* __restrict__ xyz,
                                                   int* __restrict__ fps_idx,
                                                   float* __restrict__ new_xyz) {
    const int b = blockIdx.x;
    const float* X = xyz + (size_t)b * N_ * 3;
    const int t = threadIdx.x;
    const int lane = t & 31, wave = t >> 5;

    float px[4], py[4], pz[4], dmin[4];
#pragma unroll
    for (int k = 0; k < 4; ++k) {
        int p = t + k * 1024;
        px[k] = X[p * 3 + 0];
        py[k] = X[p * 3 + 1];
        pz[k] = X[p * 3 + 2];
        dmin[k] = 1e10f;
    }

    __shared__ float s_c[3];
    __shared__ float s_val[32];
    __shared__ int   s_idx[32];
    __shared__ int   s_far;

    int far = 0;
    for (int it = 0; it < S_; ++it) {
        if (t == 0) {
            fps_idx[b * S_ + it] = far;
            float cx = X[far * 3 + 0], cy = X[far * 3 + 1], cz = X[far * 3 + 2];
            s_c[0] = cx; s_c[1] = cy; s_c[2] = cz;
            new_xyz[(b * S_ + it) * 3 + 0] = cx;
            new_xyz[(b * S_ + it) * 3 + 1] = cy;
            new_xyz[(b * S_ + it) * 3 + 2] = cz;
        }
        __syncthreads();
        const float cx = s_c[0], cy = s_c[1], cz = s_c[2];

        float best = -1.0f;
        int bidx = 0;
#pragma unroll
        for (int k = 0; k < 4; ++k) {
            float dx = px[k] - cx, dy = py[k] - cy, dz = pz[k] - cz;
            float d = dx * dx + dy * dy + dz * dz;
            dmin[k] = fminf(dmin[k], d);
            int idx = t + k * 1024;
            if (dmin[k] > best) { best = dmin[k]; bidx = idx; }
        }
#pragma unroll
        for (int off = 16; off >= 1; off >>= 1) {
            float ov = __shfl_xor(best, off, 32);
            int   oi = __shfl_xor(bidx, off, 32);
            if (ov > best || (ov == best && oi < bidx)) { best = ov; bidx = oi; }
        }
        if (lane == 0) { s_val[wave] = best; s_idx[wave] = bidx; }
        __syncthreads();
        if (wave == 0) {
            best = s_val[lane];
            bidx = s_idx[lane];
#pragma unroll
            for (int off = 16; off >= 1; off >>= 1) {
                float ov = __shfl_xor(best, off, 32);
                int   oi = __shfl_xor(bidx, off, 32);
                if (ov > best || (ov == best && oi < bidx)) { best = ov; bidx = oi; }
            }
            if (lane == 0) s_far = bidx;
        }
        __syncthreads();
        far = s_far;
    }
}

// ---------------------------------------------------------------------------
// K2: ball query. One wave32 per center; ballot+prefix, in index order.
// ---------------------------------------------------------------------------
__global__ __launch_bounds__(256) void ballquery_kernel(const float* __restrict__ xyz,
                                                        const float* __restrict__ new_xyz,
                                                        int* __restrict__ gidx) {
    const int wid = (blockIdx.x * blockDim.x + threadIdx.x) >> 5;
    const int lane = threadIdx.x & 31;
    if (wid >= B_ * S_) return;
    const int b = wid >> 10;
    const float* X = xyz + (size_t)b * N_ * 3;
    const float cx = new_xyz[wid * 3 + 0];
    const float cy = new_xyz[wid * 3 + 1];
    const float cz = new_xyz[wid * 3 + 2];
    int* out = gidx + (size_t)wid * NS_;

    int cnt = 0, first = -1;
    for (int c0 = 0; c0 < N_ && cnt < NS_; c0 += 32) {
        int p = c0 + lane;
        float dx = X[p * 3 + 0] - cx;
        float dy = X[p * 3 + 1] - cy;
        float dz = X[p * 3 + 2] - cz;
        float d2 = dx * dx + dy * dy + dz * dz;
        bool in = (d2 <= RAD2_);
        unsigned mask = (unsigned)__ballot(in);
        if (first < 0 && mask) first = c0 + __ffs(mask) - 1;
        int pre = __popc(mask & ((1u << lane) - 1u));
        if (in && (cnt + pre) < NS_) out[cnt + pre] = p;
        cnt += __popc(mask);
    }
    if (cnt > NS_) cnt = NS_;
    if (first < 0) first = 0;
    if (lane >= cnt) out[lane] = first;
}

// ---------------------------------------------------------------------------
// K3: feats = points @ W1 + b1 (f16 out).  M=32768, K=64, N=128.
// B fragments: contiguous v16h loads from pre-transposed f16 W1T.
// ---------------------------------------------------------------------------
__global__ __launch_bounds__(256) void gemm1_kernel(const float* __restrict__ points,
                                                    const _Float16* __restrict__ W1T,
                                                    const float* __restrict__ b1,
                                                    _Float16* __restrict__ feats) {
    const int wave = threadIdx.x >> 5;
    const int lane = threadIdx.x & 31;
    const int half = lane >> 4;
    const int l15  = lane & 15;

    const int mTile = blockIdx.x * 32 + (wave >> 2) * 16;
    const int nBase = (wave & 3) * 32;

    v8f acc0 = {}; v8f acc1 = {};
    const int row = mTile + l15;
    const float* ap = points + (size_t)row * CIN_;

#pragma unroll
    for (int kk = 0; kk < 2; ++kk) {
        v16h a;
        const float* ak = ap + kk * 32;
#pragma unroll
        for (int j = 0; j < 8; ++j) {
            a[j]     = (_Float16)ak[half * 16 + j];
            a[8 + j] = (_Float16)ak[half * 16 + 8 + j];
        }
        const int kbase = kk * 32 + half * 16;
#pragma unroll
        for (int nt = 0; nt < 2; ++nt) {
            const int col = nBase + nt * 16 + l15;
            const v16h bf = *(const v16h*)(W1T + (size_t)col * CIN_ + kbase);
            if (nt == 0)
                acc0 = __builtin_amdgcn_wmma_f32_16x16x32_f16(false, a, false, bf,
                                                              (short)0, acc0, false, false);
            else
                acc1 = __builtin_amdgcn_wmma_f32_16x16x32_f16(false, a, false, bf,
                                                              (short)0, acc1, false, false);
        }
    }
#pragma unroll
    for (int nt = 0; nt < 2; ++nt) {
        const int col = nBase + nt * 16 + l15;
        const float bias = b1[col];
        const v8f acc = (nt == 0) ? acc0 : acc1;
#pragma unroll
        for (int r = 0; r < 8; ++r) {
            int orow = mTile + half * 8 + r;
            feats[(size_t)orow * CMLP_ + col] = (_Float16)(acc[r] + bias);
        }
    }
}

// ---------------------------------------------------------------------------
// K4: per-(b,s) group GEMM: (32 x 128 gathered f16) @ Wc(128x256) + bc.
// Gather uses CDNA5 async direct-to-LDS loads (ASYNCcnt path).
//   mode 0: accumulate per-channel sum/sumsq.  mode 1: BN+ReLU+max -> out.
// ---------------------------------------------------------------------------
__global__ __launch_bounds__(256) void gemm2_kernel(const _Float16* __restrict__ feats,
                                                    const int* __restrict__ gidx,
                                                    const _Float16* __restrict__ WcT,
                                                    const float* __restrict__ bc,
                                                    const float* __restrict__ scale,
                                                    const float* __restrict__ shift,
                                                    float* __restrict__ gsum,
                                                    float* __restrict__ gsq,
                                                    float* __restrict__ out,
                                                    int mode) {
    const int group = blockIdx.x;            // b*S + s
    const int b = group >> 10;
    const int tid = threadIdx.x;
    const int wave = tid >> 5, lane = tid & 31;
    const int half = lane >> 4, l15 = lane & 15;

    __shared__ _Float16 sA[32 * LDSA_STRIDE];
    __shared__ int sIdx[32];
    __shared__ float sSum[COUT_];
    __shared__ float sSq[COUT_];

    if (mode == 0) { sSum[tid] = 0.0f; sSq[tid] = 0.0f; }
    if (tid < 32) sIdx[tid] = gidx[(size_t)group * NS_ + tid];
    __syncthreads();

    // async gather of 32 rows x 256B into LDS: 2x b128 per thread
    {
        const int j  = tid >> 3;           // row 0..31
        const int c0 = (tid & 7) * 16;     // 16 f16 = 32 B per thread
        const int row = sIdx[j];
        const _Float16* src = feats + ((size_t)b * N_ + row) * CMLP_ + c0;
        _Float16* dst = sA + j * LDSA_STRIDE + c0;
        unsigned lds0 = (unsigned)(uintptr_t)dst;          // flat-LDS low 32 bits
        unsigned long long g0 = (unsigned long long)(uintptr_t)src;
        asm volatile("global_load_async_to_lds_b128 %0, %1, off"
                     :: "v"(lds0), "v"(g0) : "memory");
        asm volatile("global_load_async_to_lds_b128 %0, %1, off offset:16"
                     :: "v"(lds0), "v"(g0) : "memory");
    }
    asm volatile("s_wait_asynccnt 0x0" ::: "memory");
    __syncthreads();

    const _Float16* ar0 = sA + l15 * LDSA_STRIDE;          // M-tile 0 row
    const _Float16* ar1 = sA + (16 + l15) * LDSA_STRIDE;   // M-tile 1 row

#pragma unroll
    for (int nt = 0; nt < 2; ++nt) {
        const int col = (wave * 2 + nt) * 16 + l15;
        v8f acc0 = {}; v8f acc1 = {};
#pragma unroll
        for (int kk = 0; kk < 4; ++kk) {
            const int kbase = kk * 32 + half * 16;
            const v16h bf = *(const v16h*)(WcT + (size_t)col * CMLP_ + kbase);
            const _Float16* ak0 = ar0 + kk * 32;
            const _Float16* ak1 = ar1 + kk * 32;
            v8h lo0 = *(const v8h*)(ak0 + half * 16);
            v8h hi0 = *(const v8h*)(ak0 + half * 16 + 8);
            v8h lo1 = *(const v8h*)(ak1 + half * 16);
            v8h hi1 = *(const v8h*)(ak1 + half * 16 + 8);
            v16h a0 = __builtin_shufflevector(lo0, hi0, 0, 1, 2, 3, 4, 5, 6, 7,
                                              8, 9, 10, 11, 12, 13, 14, 15);
            v16h a1 = __builtin_shufflevector(lo1, hi1, 0, 1, 2, 3, 4, 5, 6, 7,
                                              8, 9, 10, 11, 12, 13, 14, 15);
            acc0 = __builtin_amdgcn_wmma_f32_16x16x32_f16(false, a0, false, bf,
                                                          (short)0, acc0, false, false);
            acc1 = __builtin_amdgcn_wmma_f32_16x16x32_f16(false, a1, false, bf,
                                                          (short)0, acc1, false, false);
        }
        const float bias = bc[col];
        if (mode == 0) {
            float s = 0.0f, q = 0.0f;
#pragma unroll
            for (int r = 0; r < 8; ++r) {
                float h0 = acc0[r] + bias; s += h0; q += h0 * h0;
                float h1 = acc1[r] + bias; s += h1; q += h1 * h1;
            }
            atomicAdd(&sSum[col], s);
            atomicAdd(&sSq[col], q);
        } else {
            const float sc = scale[col], sh = shift[col];
            float m = 0.0f;  // ReLU output >= 0
#pragma unroll
            for (int r = 0; r < 8; ++r) {
                float h0 = fmaxf(0.0f, (acc0[r] + bias) * sc + sh);
                float h1 = fmaxf(0.0f, (acc1[r] + bias) * sc + sh);
                m = fmaxf(m, fmaxf(h0, h1));
            }
            m = fmaxf(m, __shfl_xor(m, 16, 32));  // partner lane: same channel
            if (half == 0) out[(size_t)group * COUT_ + col] = m;
        }
    }

    if (mode == 0) {
        __syncthreads();
        atomicAdd(&gsum[tid], sSum[tid]);
        atomicAdd(&gsq[tid], sSq[tid]);
    }
}

// ---------------------------------------------------------------------------
// K5: fold sums into per-channel scale/shift: y = h*scale + shift
// ---------------------------------------------------------------------------
__global__ void bn_finalize_kernel(const float* __restrict__ gsum,
                                   const float* __restrict__ gsq,
                                   const float* __restrict__ gamma,
                                   const float* __restrict__ beta,
                                   float* __restrict__ scale,
                                   float* __restrict__ shift) {
    const int c = threadIdx.x;
    const float inv = 1.0f / (float)NSAMPLES_TOTAL;
    float mean = gsum[c] * inv;
    float var  = gsq[c] * inv - mean * mean;
    float sc   = gamma[c] * rsqrtf(var + 1e-5f);
    scale[c] = sc;
    shift[c] = beta[c] - mean * sc;
}

// ---------------------------------------------------------------------------
extern "C" void kernel_launch(void* const* d_in, const int* in_sizes, int n_in,
                              void* d_out, int out_size, void* d_ws, size_t ws_size,
                              hipStream_t stream) {
    (void)in_sizes; (void)n_in; (void)out_size; (void)ws_size;
    const float* xyz    = (const float*)d_in[0];
    // d_in[1] = t is unused by the reference
    const float* points = (const float*)d_in[2];
    const float* W1     = (const float*)d_in[3];
    const float* b1     = (const float*)d_in[4];
    const float* Wc     = (const float*)d_in[5];
    const float* bc     = (const float*)d_in[6];
    const float* gamma  = (const float*)d_in[7];
    const float* beta   = (const float*)d_in[8];
    float* out = (float*)d_out;

    char* ws = (char*)d_ws;
    int*      fps_idx = (int*)(ws);                      // 8192 ints
    float*    new_xyz = (float*)(ws + 32768);            // 24576 f32
    int*      gidx    = (int*)(ws + 131072);             // 262144 ints
    _Float16* feats   = (_Float16*)(ws + 1179648);       // 4194304 f16 (8 MB)
    float*    gsum    = (float*)(ws + 9568256);          // 256
    float*    gsq     = gsum + 256;                      // 256
    float*    scale   = gsum + 512;                      // 256
    float*    shift   = gsum + 768;                      // 256
    _Float16* W1T     = (_Float16*)(ws + 9572352);       // 8192 f16
    _Float16* WcT     = (_Float16*)(ws + 9588736);       // 32768 f16

    init_stats_kernel<<<1, 512, 0, stream>>>(gsum);
    prep_weights_kernel<<<128, 256, 0, stream>>>(W1, Wc, W1T, WcT);
    fps_kernel<<<B_, 1024, 0, stream>>>(xyz, fps_idx, new_xyz);
    ballquery_kernel<<<(B_ * S_) / 8, 256, 0, stream>>>(xyz, new_xyz, gidx);
    gemm1_kernel<<<(B_ * N_) / 32, 256, 0, stream>>>(points, W1T, b1, feats);
    gemm2_kernel<<<B_ * S_, 256, 0, stream>>>(feats, gidx, WcT, bc, scale, shift,
                                              gsum, gsq, out, 0);
    bn_finalize_kernel<<<1, 256, 0, stream>>>(gsum, gsq, gamma, beta, scale, shift);
    gemm2_kernel<<<B_ * S_, 256, 0, stream>>>(feats, gidx, WcT, bc, scale, shift,
                                              gsum, gsq, out, 1);
}
